// S5_three_to_one_80152679678392
// MI455X (gfx1250) — matrix operand order repaired and checked by reading, not compiled, not run on hardware
//
#include <hip/hip_runtime.h>
#include <hip/hip_bf16.h>
#include <stdint.h>

typedef __attribute__((ext_vector_type(16))) _Float16 v16h;
typedef __attribute__((ext_vector_type(8)))  float    v8f;

#define THREADS 128
#define LSEQ    16384
#define CHUNK   (LSEQ / THREADS)            // 128 timesteps per thread

// ---- LDS layout (float offsets) ----
#define P_LX      0                          // 16384 floats: input row
#define P_LY      16384                      // 49152 floats: 3-channel concat
#define P_SCRATCH (16384 + 3*16384)          // 9216 floats: scan carries / WMMA staging
#define SCRATCH_F 9216
#define P_PARAMS  (P_SCRATCH + SCRATCH_F)    // 512 floats: derived block params
#define PARAMS_F  512
#define SMEM_F    (P_PARAMS + PARAMS_F)      // 75264 floats = 294 KB (<320 KB/WGP)

// ---- param area offsets (sized for P=18,H=3) ----
#define PA_ABR 0      // Lambda_bar re [18]
#define PA_ABI 18     // Lambda_bar im [18]
#define PA_BBR 36     // B_bar re [18*3]
#define PA_BBI 90     // B_bar im [18*3]
#define PA_CB  144    // 64 x 4 canonical WMMA-B source (2*C_re / -2*C_im), zero padded
#define PA_D   400
#define PA_L1W 403
#define PA_L1B 406
#define PA_L2W 409
#define PA_L2B 412
#define PA_ENC 415    // 2H*H <= 18
#define PA_DEC 433    // H*H  <= 9

struct KParams { const float* p[64]; };

__device__ __forceinline__ float gelu_tanh(float x) {
    float x3 = x * x * x;
    return 0.5f * x * (1.0f + tanhf(0.7978845608028654f * (x + 0.044715f * x3)));
}

// one fused S5 block (LN1 -> SSM scan -> gelu+res -> LN2 -> GEGLU FFN -> res)
template <int H, int P, bool FINAL>
__device__ __forceinline__ void s5_block(const float* const* prm, float* smem,
                                         const float* in_seq, int in_stride,
                                         float* out_seq, int out_stride,
                                         float* gout,
                                         const float* l3w_g, const float* l3b_g) {
    const int tid  = threadIdx.x;
    const int lane = tid & 31;
    const int wave = tid >> 5;
    float* PAR = smem + P_PARAMS;
    float* SC  = smem + P_SCRATCH;

    // ---- derive discretized SSM params (ZOH) into LDS ----
    if (tid == 0) {
        const float* Lre = prm[0]; const float* Lim = prm[1]; const float* ls = prm[2];
        const float* Bre = prm[3]; const float* Bim = prm[4];
        const float* Cre = prm[5]; const float* Cim = prm[6];
        for (int i = 0; i < 64 * 4; ++i) PAR[PA_CB + i] = 0.0f;
        for (int p = 0; p < P; ++p) {
            float dt  = expf(ls[p]);
            float er  = expf(Lre[p] * dt);
            float th  = Lim[p] * dt;
            float abr = er * cosf(th);
            float abi = er * sinf(th);
            PAR[PA_ABR + p] = abr; PAR[PA_ABI + p] = abi;
            float dr = Lre[p], di = Lim[p];
            float inv = 1.0f / (dr * dr + di * di);
            float nr = abr - 1.0f, ni = abi;
            float wr = (nr * dr + ni * di) * inv;
            float wi = (ni * dr - nr * di) * inv;
            for (int h = 0; h < H; ++h) {
                float br = Bre[p * H + h], bi = Bim[p * H + h];
                PAR[PA_BBR + p * H + h] = wr * br - wi * bi;
                PAR[PA_BBI + p * H + h] = wr * bi + wi * br;
            }
            for (int h = 0; h < H; ++h) {   // WMMA B source: y = 2*Re(x*conj-less C)
                PAR[PA_CB + (2 * p) * 4 + h]     =  2.0f * Cre[h * P + p];
                PAR[PA_CB + (2 * p + 1) * 4 + h] = -2.0f * Cim[h * P + p];
            }
        }
        for (int h = 0; h < H; ++h) {
            PAR[PA_D   + h] = prm[7][h];
            PAR[PA_L1W + h] = prm[8][h];  PAR[PA_L1B + h] = prm[9][h];
            PAR[PA_L2W + h] = prm[10][h]; PAR[PA_L2B + h] = prm[11][h];
        }
        for (int i = 0; i < 2 * H * H; ++i) PAR[PA_ENC + i] = prm[12][i];
        for (int i = 0; i < H * H;     ++i) PAR[PA_DEC + i] = prm[13][i];
    }
    __syncthreads();

    // cache per-state params in registers (fully unrolled, template P/H)
    float abr[P], abi[P], bbr[P][H], bbi[P][H];
#pragma unroll
    for (int p = 0; p < P; ++p) {
        abr[p] = PAR[PA_ABR + p]; abi[p] = PAR[PA_ABI + p];
#pragma unroll
        for (int h = 0; h < H; ++h) {
            bbr[p][h] = PAR[PA_BBR + p * H + h];
            bbi[p][h] = PAR[PA_BBI + p * H + h];
        }
    }
    float l1w[H], l1b[H], l2w[H], l2b[H], Dh[H];
#pragma unroll
    for (int h = 0; h < H; ++h) {
        l1w[h] = PAR[PA_L1W + h]; l1b[h] = PAR[PA_L1B + h];
        l2w[h] = PAR[PA_L2W + h]; l2b[h] = PAR[PA_L2B + h];
        Dh[h]  = PAR[PA_D + h];
    }

    auto ln1 = [&](int t, float* fx) {
        float u[H];
#pragma unroll
        for (int h = 0; h < H; ++h) u[h] = in_seq[t * in_stride + h];
        float mu = 0.f;
#pragma unroll
        for (int h = 0; h < H; ++h) mu += u[h];
        mu *= (1.0f / H);
        float var = 0.f;
#pragma unroll
        for (int h = 0; h < H; ++h) { float d = u[h] - mu; var += d * d; }
        var *= (1.0f / H);
        float r = rsqrtf(var + 1e-5f);
#pragma unroll
        for (int h = 0; h < H; ++h) fx[h] = (u[h] - mu) * r * l1w[h] + l1b[h];
    };

    const int t0 = tid * CHUNK;

    // ---- Pass A: per-chunk (A,b) reduction ----
    float cAr[P], cAi[P], cbr[P], cbi[P];
#pragma unroll
    for (int p = 0; p < P; ++p) { cAr[p] = 1.f; cAi[p] = 0.f; cbr[p] = 0.f; cbi[p] = 0.f; }
    for (int j = 0; j < CHUNK; ++j) {
        int t = t0 + j;
        float fx[H]; ln1(t, fx);
#pragma unroll
        for (int p = 0; p < P; ++p) {
            float bur = 0.f, bui = 0.f;
#pragma unroll
            for (int h = 0; h < H; ++h) { bur += fx[h] * bbr[p][h]; bui += fx[h] * bbi[p][h]; }
            float nbr = abr[p] * cbr[p] - abi[p] * cbi[p] + bur;
            float nbi = abr[p] * cbi[p] + abi[p] * cbr[p] + bui;
            cbr[p] = nbr; cbi[p] = nbi;
            float nAr = abr[p] * cAr[p] - abi[p] * cAi[p];
            float nAi = abr[p] * cAi[p] + abi[p] * cAr[p];
            cAr[p] = nAr; cAi[p] = nAi;
        }
    }
#pragma unroll
    for (int p = 0; p < P; ++p) {
        SC[tid * 4 * P + 4 * p + 0] = cAr[p]; SC[tid * 4 * P + 4 * p + 1] = cAi[p];
        SC[tid * 4 * P + 4 * p + 2] = cbr[p]; SC[tid * 4 * P + 4 * p + 3] = cbi[p];
    }
    __syncthreads();

    // ---- cross-thread Hillis-Steele scan of carries ----
    for (int off = 1; off < THREADS; off <<= 1) {
        float tAr[P], tAi[P], tbr[P], tbi[P];
        bool act = (tid >= off);
        if (act) {
#pragma unroll
            for (int p = 0; p < P; ++p) {
                tAr[p] = SC[(tid - off) * 4 * P + 4 * p + 0];
                tAi[p] = SC[(tid - off) * 4 * P + 4 * p + 1];
                tbr[p] = SC[(tid - off) * 4 * P + 4 * p + 2];
                tbi[p] = SC[(tid - off) * 4 * P + 4 * p + 3];
            }
        }
        __syncthreads();
        if (act) {
#pragma unroll
            for (int p = 0; p < P; ++p) {
                float nAr = cAr[p] * tAr[p] - cAi[p] * tAi[p];
                float nAi = cAr[p] * tAi[p] + cAi[p] * tAr[p];
                float nbr = cAr[p] * tbr[p] - cAi[p] * tbi[p] + cbr[p];
                float nbi = cAr[p] * tbi[p] + cAi[p] * tbr[p] + cbi[p];
                cAr[p] = nAr; cAi[p] = nAi; cbr[p] = nbr; cbi[p] = nbi;
                SC[tid * 4 * P + 4 * p + 0] = nAr; SC[tid * 4 * P + 4 * p + 1] = nAi;
                SC[tid * 4 * P + 4 * p + 2] = nbr; SC[tid * 4 * P + 4 * p + 3] = nbi;
            }
        }
        __syncthreads();
    }
    // exclusive prefix -> initial state for this chunk
    float sr[P], si[P];
#pragma unroll
    for (int p = 0; p < P; ++p) { sr[p] = 0.f; si[p] = 0.f; }
    if (tid > 0) {
#pragma unroll
        for (int p = 0; p < P; ++p) {
            sr[p] = SC[(tid - 1) * 4 * P + 4 * p + 2];
            si[p] = SC[(tid - 1) * 4 * P + 4 * p + 3];
        }
    }
    __syncthreads();   // scratch region now free -> reuse as WMMA staging

    // ---- WMMA staging (reuses scan scratch) ----
    constexpr int KCH = (2 * P + 31) / 32;               // K chunks of 32 (2P reals)
    _Float16* aS = (_Float16*)SC + wave * (32 * 64);     // 32 rows x 64 f16 per wave
    float*    dS = SC + 4096 + wave * (32 * 4);          // 32 rows x 4 f32 per wave
    for (int i = lane; i < 32 * 64; i += 32) aS[i] = (_Float16)0.0f;

    // constant B fragment: 32x16 f16 layout (lanes 0-15 K=0..15, lanes 16-31 K=16..31)
    v16h bfr[KCH];
    {
        int n  = lane & 15;
        int kb = (lane >= 16) ? 16 : 0;
#pragma unroll
        for (int c = 0; c < KCH; ++c) {
#pragma unroll
            for (int vi = 0; vi < 8; ++vi) {
                int k0 = 32 * c + kb + 2 * vi;
                float v0 = (n < 4) ? PAR[PA_CB + k0 * 4 + n] : 0.0f;
                float v1 = (n < 4) ? PAR[PA_CB + (k0 + 1) * 4 + n] : 0.0f;
                bfr[c][2 * vi]     = (_Float16)v0;
                bfr[c][2 * vi + 1] = (_Float16)v1;
            }
        }
    }
    float l3wr[H]; float l3b0 = 0.f;
    if constexpr (FINAL) {
#pragma unroll
        for (int h = 0; h < H; ++h) l3wr[h] = l3w_g[h];
        l3b0 = l3b_g[0];
    }
    __syncthreads();

    // ---- Pass B: re-scan + WMMA output projection + fused epilogue ----
    for (int j = 0; j < CHUNK; ++j) {
        int t = t0 + j;
        float fx[H]; ln1(t, fx);
#pragma unroll
        for (int p = 0; p < P; ++p) {
            float bur = 0.f, bui = 0.f;
#pragma unroll
            for (int h = 0; h < H; ++h) { bur += fx[h] * bbr[p][h]; bui += fx[h] * bbi[p][h]; }
            float nr = abr[p] * sr[p] - abi[p] * si[p] + bur;
            float ni = abr[p] * si[p] + abi[p] * sr[p] + bui;
            sr[p] = nr; si[p] = ni;
        }
        __syncthreads();                       // WAR on aS vs previous gather
#pragma unroll
        for (int p = 0; p < P; ++p) {          // stage this lane's state as A row
            aS[lane * 64 + 2 * p]     = (_Float16)sr[p];
            aS[lane * 64 + 2 * p + 1] = (_Float16)si[p];
        }
        __syncthreads();
        // two 16-row tiles (rows = lanes 0-15, lanes 16-31)
#pragma unroll
        for (int tI = 0; tI < 2; ++tI) {
            v8f acc = {};
#pragma unroll
            for (int c = 0; c < KCH; ++c) {
                v16h afr{};
                int m   = (lane & 15) + 16 * tI;          // staging row
                int kb2 = (lane >= 16) ? 16 : 0;
#pragma unroll
                for (int vi = 0; vi < 8; ++vi) {
                    int k = 32 * c + kb2 + 2 * vi;
                    afr[2 * vi]     = aS[m * 64 + k];
                    afr[2 * vi + 1] = aS[m * 64 + k + 1];
                }
                acc = __builtin_amdgcn_wmma_f32_16x16x32_f16(
                          false, afr, false, bfr[c], (short)0, acc, false, false);
            }
            int n = lane & 15;
            if (n < 4) {
#pragma unroll
                for (int r = 0; r < 8; ++r) {
                    int m = 16 * tI + r + ((lane >= 16) ? 8 : 0);
                    dS[m * 4 + n] = acc[r];
                }
            }
        }
        __syncthreads();
        // epilogue: y = wmma_out + D*u ; gelu+res ; LN2 ; GEGLU FFN ; res
        float g1[H];
#pragma unroll
        for (int h = 0; h < H; ++h) {
            float y = dS[lane * 4 + h] + Dh[h] * fx[h];
            g1[h] = gelu_tanh(y) + fx[h];
        }
        float mu = 0.f;
#pragma unroll
        for (int h = 0; h < H; ++h) mu += g1[h];
        mu *= (1.0f / H);
        float var = 0.f;
#pragma unroll
        for (int h = 0; h < H; ++h) { float d = g1[h] - mu; var += d * d; }
        var *= (1.0f / H);
        float rr = rsqrtf(var + 1e-5f);
        float f2[H];
#pragma unroll
        for (int h = 0; h < H; ++h) f2[h] = (g1[h] - mu) * rr * l2w[h] + l2b[h];
        float he[2 * H];
#pragma unroll
        for (int f = 0; f < 2 * H; ++f) {
            float s = 0.f;
#pragma unroll
            for (int h = 0; h < H; ++h) s += f2[h] * PAR[PA_ENC + f * H + h];
            he[f] = s;
        }
        float hh[H];
#pragma unroll
        for (int h = 0; h < H; ++h) hh[h] = he[h] * gelu_tanh(he[H + h]);
        float o[H];
#pragma unroll
        for (int h = 0; h < H; ++h) {
            float s = f2[h];
#pragma unroll
            for (int k = 0; k < H; ++k) s += PAR[PA_DEC + h * H + k] * hh[k];
            o[h] = s;
        }
        if constexpr (FINAL) {
            float ov = l3b0;
#pragma unroll
            for (int h = 0; h < H; ++h) ov += o[h] * l3wr[h];
            gout[t] = ov;
        } else {
#pragma unroll
            for (int h = 0; h < H; ++h) out_seq[t * out_stride + h] = o[h];
        }
    }
    __syncthreads();
}

__global__ __launch_bounds__(THREADS)
void s5_fused_kernel(KParams kp, const float* __restrict__ x, float* __restrict__ out) {
    extern __shared__ float smem[];
    const int b = blockIdx.x;
    const float* xrow = x + (size_t)b * LSEQ;
    float* lx = smem + P_LX;
    float* ly = smem + P_LY;

    // Phase 0: async global -> LDS staging of the input row (b128 per lane)
    for (int k = 0; k < LSEQ / (4 * THREADS); ++k) {
        int idx = (threadIdx.x + k * THREADS) * 4;
        unsigned lds = (unsigned)(uintptr_t)(lx + idx);
        const float* g = xrow + idx;
        asm volatile("global_load_async_to_lds_b128 %0, %1, off"
                     :: "v"(lds), "v"(g) : "memory");
    }
    asm volatile("s_wait_asynccnt 0x0" ::: "memory");
    __syncthreads();

    // s1/s2/s3: H=1,P=6 -> channels of ly ; s4: H=3,P=18 fused with final linear
    s5_block<1, 6,  false>(kp.p + 1,  smem, lx, 1, ly + 0, 3, nullptr, nullptr, nullptr);
    s5_block<1, 6,  false>(kp.p + 15, smem, lx, 1, ly + 1, 3, nullptr, nullptr, nullptr);
    s5_block<1, 6,  false>(kp.p + 29, smem, lx, 1, ly + 2, 3, nullptr, nullptr, nullptr);
    s5_block<3, 18, true >(kp.p + 43, smem, ly, 3, nullptr, 0,
                           out + (size_t)b * LSEQ, kp.p[57], kp.p[58]);
}

extern "C" void kernel_launch(void* const* d_in, const int* in_sizes, int n_in,
                              void* d_out, int out_size, void* d_ws, size_t ws_size,
                              hipStream_t stream) {
    (void)in_sizes; (void)d_ws; (void)ws_size; (void)out_size;
    KParams kp{};
    for (int i = 0; i < n_in && i < 64; ++i) kp.p[i] = (const float*)d_in[i];
    const float* x = (const float*)d_in[0];
    float* out = (float*)d_out;
    size_t shmem = SMEM_F * sizeof(float);
    s5_fused_kernel<<<dim3(64), dim3(THREADS), shmem, stream>>>(kp, x, out);
}
